// QATBNN_61272003445437
// MI455X (gfx1250) — 8-bit matrix operand order repaired and checked by reading, compiled, not run on hardware
//
#include <hip/hip_runtime.h>

typedef __attribute__((ext_vector_type(8))) int v8i;

#define NB     4096
#define K2     144
#define K3     288
#define KF     3872
#define KFP    3904
#define KFT    61

#define OFF_ACT2 0ull
#define OFF_ACT4 0ull
#define OFF_ACT1 75497472ull
#define OFF_ACT3 75497472ull
#define OFF_B2   138936320ull
#define OFF_B3   138942464ull
#define OFF_BF   138952704ull

__device__ __forceinline__ int sgnf(float v){ return v > 0.f ? 1 : (v < 0.f ? -1 : 0); }
__device__ __forceinline__ int sgni(int v){ return v > 0 ? 1 : (v < 0 ? -1 : 0); }

__device__ __forceinline__ int kbaseB(int v, int hi){ return ((v >> 1) << 4) + ((v & 1) << 2) + (hi << 3); }

__global__ void pack_w2(const float* __restrict__ w2, unsigned* __restrict__ bout){
  int idx = blockIdx.x * 256 + threadIdx.x;
  int v = idx & 7, lane = (idx >> 3) & 31, t = idx >> 8;
  int kt = t % 3, nt = t / 3;
  int hi = lane >> 4, n = nt * 16 + (lane & 15);
  unsigned d = 0;
  #pragma unroll
  for (int j = 0; j < 4; ++j){
    int k = kt * 64 + kbaseB(v, hi) + j;
    int b = 0;
    if (k < K2){
      int tap = k >> 4, cin = k & 15, kh = tap / 3, kw = tap % 3;
      b = sgnf(w2[((n * 16 + cin) * 3 + kh) * 3 + kw]);
    }
    d |= (unsigned)(b & 0xFF) << (8 * j);
  }
  bout[idx] = d;
}

__global__ void pack_w3(const float* __restrict__ w3, unsigned* __restrict__ bout){
  int idx = blockIdx.x * 256 + threadIdx.x;
  int v = idx & 7, lane = (idx >> 3) & 31, t = idx >> 8;
  int kt = t % 5, nt = t / 5;
  int hi = lane >> 4, n = nt * 16 + (lane & 15);
  unsigned d = 0;
  #pragma unroll
  for (int j = 0; j < 4; ++j){
    int k = kt * 64 + kbaseB(v, hi) + j;
    int b = 0;
    if (k < K3){
      int tap = k >> 5, cin = k & 31, kh = tap / 3, kw = tap % 3;
      b = sgnf(w3[((n * 32 + cin) * 3 + kh) * 3 + kw]);
    }
    d |= (unsigned)(b & 0xFF) << (8 * j);
  }
  bout[idx] = d;
}

__global__ void pack_wf(const float* __restrict__ wf, unsigned* __restrict__ bout){
  int idx = blockIdx.x * 256 + threadIdx.x;
  int v = idx & 7, lane = (idx >> 3) & 31, kt = idx >> 8;
  int hi = lane >> 4, n = lane & 15;
  unsigned d = 0;
  #pragma unroll
  for (int j = 0; j < 4; ++j){
    int k = kt * 64 + kbaseB(v, hi) + j;
    int b = (n < 10 && k < KF) ? sgnf(wf[n * KF + k]) : 0;
    d |= (unsigned)(b & 0xFF) << (8 * j);
  }
  bout[idx] = d;
}

__global__ void conv1_kernel(const float* __restrict__ x, const float* __restrict__ w1,
                             signed char* __restrict__ act1){
  __shared__ int ws1[144];
  int tid = threadIdx.x;
  if (tid < 144) ws1[tid] = sgnf(w1[tid]);
  __syncthreads();
  int idx = blockIdx.x * 256 + tid;
  int n = idx / 676, p = idx % 676;
  int oh = p / 26, ow = p % 26;
  const float* xp = x + n * 784 + oh * 28 + ow;
  int xb[9];
  #pragma unroll
  for (int kh = 0; kh < 3; ++kh)
    #pragma unroll
    for (int kw = 0; kw < 3; ++kw)
      xb[kh * 3 + kw] = (xp[kh * 28 + kw] > 0.5f) ? 1 : -1;
  union { int4 v; signed char c[16]; } u;
  #pragma unroll
  for (int oc = 0; oc < 16; ++oc){
    int acc = 0;
    #pragma unroll
    for (int t = 0; t < 9; ++t) acc += xb[t] * ws1[oc * 9 + t];
    u.c[oc] = (signed char)sgni(acc);
  }
  ((int4*)act1)[idx] = u.v;
}

__device__ __forceinline__ int kbaseA(int v, int hi){ return ((v >> 1) << 4) + ((v & 1) << 2) + (hi << 3); }

__global__ void __launch_bounds__(256) conv2_kernel(const signed char* __restrict__ act1,
                                                    const unsigned* __restrict__ b2,
                                                    signed char* __restrict__ act2){
  const int lane = threadIdx.x & 31, wave = threadIdx.x >> 5;
  const int st = blockIdx.x * 8 + wave;
  const int hi = lane >> 4, nlo = lane & 15;
  const int g0 = st * 32;
  const int* a1i = (const int*)act1;
  const v8i* bf = (const v8i*)b2;

  const int gA0 = g0 + nlo, gA1 = g0 + 16 + nlo;
  const int img = gA0 / 576;
  const int pix0 = gA0 % 576, pix1 = gA1 % 576;
  const int oh0 = pix0 / 24, ow0 = pix0 % 24;
  const int oh1 = pix1 / 24, ow1 = pix1 % 24;

  v8i acc00 = {}, acc01 = {}, acc10 = {}, acc11 = {};
  #pragma unroll
  for (int kt = 0; kt < 3; ++kt){
    v8i a0, a1;
    #pragma unroll
    for (int v = 0; v < 8; ++v){
      int k = kt * 64 + kbaseA(v, hi);
      int tap = k >> 4, c0 = k & 15;
      int v0 = 0, v1 = 0;
      if (tap < 9){
        int kh = tap / 3, kw = tap % 3;
        v0 = a1i[((img * 26 + oh0 + kh) * 26 + (ow0 + kw)) * 4 + (c0 >> 2)];
        v1 = a1i[((img * 26 + oh1 + kh) * 26 + (ow1 + kw)) * 4 + (c0 >> 2)];
      }
      a0[v] = v0; a1[v] = v1;
    }
    v8i b0 = bf[(0 * 3 + kt) * 32 + lane];
    v8i b1 = bf[(1 * 3 + kt) * 32 + lane];
    acc00 = __builtin_amdgcn_wmma_i32_16x16x64_iu8(true, a0, true, b0, acc00, false, false);
    acc01 = __builtin_amdgcn_wmma_i32_16x16x64_iu8(true, a0, true, b1, acc01, false, false);
    acc10 = __builtin_amdgcn_wmma_i32_16x16x64_iu8(true, a1, true, b0, acc10, false, false);
    acc11 = __builtin_amdgcn_wmma_i32_16x16x64_iu8(true, a1, true, b1, acc11, false, false);
  }
  #pragma unroll
  for (int d = 0; d < 8; ++d){
    int gm0 = g0 + d + hi * 8;
    int gm1 = g0 + 16 + d + hi * 8;
    act2[gm0 * 32 + nlo]      = (signed char)sgni(acc00[d]);
    act2[gm0 * 32 + 16 + nlo] = (signed char)sgni(acc01[d]);
    act2[gm1 * 32 + nlo]      = (signed char)sgni(acc10[d]);
    act2[gm1 * 32 + 16 + nlo] = (signed char)sgni(acc11[d]);
  }
}

__global__ void __launch_bounds__(256) conv3_kernel(const signed char* __restrict__ act2,
                                                    const unsigned* __restrict__ b3,
                                                    signed char* __restrict__ act3){
  const int lane = threadIdx.x & 31, wave = threadIdx.x >> 5;
  const int st = blockIdx.x * 8 + wave;
  const int hi = lane >> 4, nlo = lane & 15;
  const int g0 = st * 32;
  const int* a2i = (const int*)act2;
  const v8i* bf = (const v8i*)b3;

  const int gA0 = g0 + nlo, gA1 = g0 + 16 + nlo;
  const int img0 = gA0 / 484, pix0 = gA0 % 484;
  const int img1 = gA1 / 484, pix1 = gA1 % 484;
  const int oh0 = pix0 / 22, ow0 = pix0 % 22;
  const int oh1 = pix1 / 22, ow1 = pix1 % 22;

  v8i acc00 = {}, acc01 = {}, acc10 = {}, acc11 = {};
  #pragma unroll
  for (int kt = 0; kt < 5; ++kt){
    v8i a0, a1;
    #pragma unroll
    for (int v = 0; v < 8; ++v){
      int k = kt * 64 + kbaseA(v, hi);
      int tap = k >> 5, c0 = k & 31;
      int v0 = 0, v1 = 0;
      if (tap < 9){
        int kh = tap / 3, kw = tap % 3;
        v0 = a2i[((img0 * 24 + oh0 + kh) * 24 + (ow0 + kw)) * 8 + (c0 >> 2)];
        v1 = a2i[((img1 * 24 + oh1 + kh) * 24 + (ow1 + kw)) * 8 + (c0 >> 2)];
      }
      a0[v] = v0; a1[v] = v1;
    }
    v8i b0 = bf[(0 * 5 + kt) * 32 + lane];
    v8i b1 = bf[(1 * 5 + kt) * 32 + lane];
    acc00 = __builtin_amdgcn_wmma_i32_16x16x64_iu8(true, a0, true, b0, acc00, false, false);
    acc01 = __builtin_amdgcn_wmma_i32_16x16x64_iu8(true, a0, true, b1, acc01, false, false);
    acc10 = __builtin_amdgcn_wmma_i32_16x16x64_iu8(true, a1, true, b0, acc10, false, false);
    acc11 = __builtin_amdgcn_wmma_i32_16x16x64_iu8(true, a1, true, b1, acc11, false, false);
  }
  #pragma unroll
  for (int d = 0; d < 8; ++d){
    int gm0 = g0 + d + hi * 8;
    int gm1 = g0 + 16 + d + hi * 8;
    act3[gm0 * 32 + nlo]      = (signed char)sgni(acc00[d]);
    act3[gm0 * 32 + 16 + nlo] = (signed char)sgni(acc01[d]);
    act3[gm1 * 32 + nlo]      = (signed char)sgni(acc10[d]);
    act3[gm1 * 32 + 16 + nlo] = (signed char)sgni(acc11[d]);
  }
}

__global__ void pool_kernel(const signed char* __restrict__ act3, signed char* __restrict__ act4){
  int idx = blockIdx.x * 256 + threadIdx.x;
  int n = idx / KFP, col = idx % KFP;
  int out = 0;
  if (col < KF){
    int c = col / 121, hw = col % 121, h = hw / 11, w = hw % 11;
    const signed char* p = act3 + n * 484 * 32;
    int s = p[((2 * h) * 22 + 2 * w) * 32 + c]
          + p[((2 * h) * 22 + 2 * w + 1) * 32 + c]
          + p[((2 * h + 1) * 22 + 2 * w) * 32 + c]
          + p[((2 * h + 1) * 22 + 2 * w + 1) * 32 + c];
    out = sgni(s);
  }
  act4[idx] = (signed char)out;
}

__global__ void __launch_bounds__(256) fc_kernel(const signed char* __restrict__ act4,
                                                 const unsigned* __restrict__ bfp,
                                                 const float* __restrict__ bias,
                                                 float* __restrict__ out){
  __shared__ int lg[8][16][16];
  const int lane = threadIdx.x & 31, wave = threadIdx.x >> 5;
  const int tile = blockIdx.x * 8 + wave;
  const int hi = lane >> 4, nlo = lane & 15;
  const int row = tile * 16 + nlo;
  const int* a4 = (const int*)act4;
  const v8i* bf = (const v8i*)bfp;
  v8i accA = {}, accB = {};
  for (int kt = 0; kt < KFT - 1; kt += 2){
    v8i a0, a1;
    #pragma unroll
    for (int v = 0; v < 8; ++v){
      int k0 = kt * 64 + kbaseA(v, hi);
      a0[v] = a4[row * (KFP / 4) + (k0 >> 2)];
      a1[v] = a4[row * (KFP / 4) + ((k0 + 64) >> 2)];
    }
    v8i b0 = bf[kt * 32 + lane];
    v8i b1 = bf[(kt + 1) * 32 + lane];
    accA = __builtin_amdgcn_wmma_i32_16x16x64_iu8(true, a0, true, b0, accA, false, false);
    accB = __builtin_amdgcn_wmma_i32_16x16x64_iu8(true, a1, true, b1, accB, false, false);
  }
  {
    v8i a;
    #pragma unroll
    for (int v = 0; v < 8; ++v){
      int k = (KFT - 1) * 64 + kbaseA(v, hi);
      a[v] = a4[row * (KFP / 4) + (k >> 2)];
    }
    v8i b = bf[(KFT - 1) * 32 + lane];
    accA = __builtin_amdgcn_wmma_i32_16x16x64_iu8(true, a, true, b, accA, false, false);
  }
  #pragma unroll
  for (int d = 0; d < 8; ++d) lg[wave][d + hi * 8][nlo] = accA[d] + accB[d];
  __syncthreads();
  if (lane < 16){
    int r = tile * 16 + lane;
    float y[10]; float mx = -1e30f;
    #pragma unroll
    for (int j = 0; j < 10; ++j){ y[j] = (float)lg[wave][lane][j] + bias[j]; mx = fmaxf(mx, y[j]); }
    float se = 0.f;
    #pragma unroll
    for (int j = 0; j < 10; ++j) se += expf(y[j] - mx);
    float lse = mx + logf(se);
    #pragma unroll
    for (int j = 0; j < 10; ++j) out[r * 10 + j] = y[j] - lse;
  }
}

extern "C" void kernel_launch(void* const* d_in, const int* in_sizes, int n_in,
                              void* d_out, int out_size, void* d_ws, size_t ws_size,
                              hipStream_t stream){
  const float* x  = (const float*)d_in[0];
  const float* w1 = (const float*)d_in[1];
  const float* w2 = (const float*)d_in[2];
  const float* w3 = (const float*)d_in[3];
  const float* wf = (const float*)d_in[4];
  const float* fb = (const float*)d_in[5];
  char* ws = (char*)d_ws;
  signed char* act1 = (signed char*)(ws + OFF_ACT1);
  signed char* act2 = (signed char*)(ws + OFF_ACT2);
  signed char* act3 = (signed char*)(ws + OFF_ACT3);
  signed char* act4 = (signed char*)(ws + OFF_ACT4);
  unsigned* b2  = (unsigned*)(ws + OFF_B2);
  unsigned* b3  = (unsigned*)(ws + OFF_B3);
  unsigned* bfp = (unsigned*)(ws + OFF_BF);

  pack_w2<<<6, 256, 0, stream>>>(w2, b2);
  pack_w3<<<10, 256, 0, stream>>>(w3, b3);
  pack_wf<<<61, 256, 0, stream>>>(wf, bfp);
  conv1_kernel<<<10816, 256, 0, stream>>>(x, w1, act1);
  conv2_kernel<<<9216, 256, 0, stream>>>(act1, b2, act2);
  conv3_kernel<<<7744, 256, 0, stream>>>(act2, b3, act3);
  pool_kernel<<<62464, 256, 0, stream>>>(act3, act4);
  fc_kernel<<<32, 256, 0, stream>>>(act4, bfp, fb, (float*)d_out);
}
